// HeteroGNN_56126632624588
// MI455X (gfx1250) — matrix operand order repaired and checked by reading, not compiled, not run on hardware
//
#include <hip/hip_runtime.h>
#include <hip/hip_bf16.h>

// CDNA5 / gfx1250 wave32 WMMA types
typedef __attribute__((ext_vector_type(16))) __bf16 v16bf;
typedef __attribute__((ext_vector_type(8)))  float  v8f;

#define HDIM 64

// ---------------------------------------------------------------------------
// zero fill
// ---------------------------------------------------------------------------
__global__ void hg_zero_f32(float* __restrict__ p, size_t n) {
    size_t i = (size_t)blockIdx.x * blockDim.x + threadIdx.x;
    size_t stride = (size_t)gridDim.x * blockDim.x;
    for (; i < n; i += stride) p[i] = 0.0f;
}

// ---------------------------------------------------------------------------
// edge scatter-add: 16 threads per edge, float4 per thread.
// agg[dst] += x_src[src]; cnt[dst] += 1 (once per edge, first layer only)
// ---------------------------------------------------------------------------
__global__ void hg_scatter_add(const float* __restrict__ xsrc,
                               const int*   __restrict__ src,
                               const int*   __restrict__ dst,
                               float* __restrict__ agg,
                               float* __restrict__ cnt,
                               int n_edges, int do_cnt) {
    long long t = (long long)blockIdx.x * blockDim.x + threadIdx.x;
    int e = (int)(t >> 4);
    if (e >= n_edges) return;
    int q = ((int)t & 15) * 4;
    int s = src[e];
    int d = dst[e];
    float4 v = *(const float4*)(xsrc + (size_t)s * HDIM + q);
    float* dp = agg + (size_t)d * HDIM + q;
    atomicAdd(dp + 0, v.x);
    atomicAdd(dp + 1, v.y);
    atomicAdd(dp + 2, v.z);
    atomicAdd(dp + 3, v.w);
    if (do_cnt && q == 0) atomicAdd(cnt + d, 1.0f);
}

// ---------------------------------------------------------------------------
// operand packing helpers (f32 -> bf16), per-lane.
//
// A (16x32 bf16, M x K): lane holds row M = lane&15.
//   lo lanes: K = kh+{0..7} U kh+{16..23} with kh = 0
//   hi lanes: same with kh = 8          (ISA 7.12.2, 16-bit A table)
//   -> caller passes p = row_ptr + ko + kh; we read p[0..7] and p[16..23].
//
// B (32x16 bf16, K x N): lane holds column N = lane&15;
//   same K order as A: lo lanes K = ko+{0..7} U ko+{16..23}, hi lanes kh = 8.
//   B[k][n] = W[n][k], W row-major; we read p[0..7] and p[16..23].
// ---------------------------------------------------------------------------
__device__ __forceinline__ v16bf hg_pack_a(const float* __restrict__ p, float s) {
    v16bf r;
#pragma unroll
    for (int j = 0; j < 8; ++j) r[j] = (__bf16)(p[j] * s);
#pragma unroll
    for (int j = 0; j < 8; ++j) r[8 + j] = (__bf16)(p[16 + j] * s);
    return r;
}

__device__ __forceinline__ v16bf hg_pack_b(const float* __restrict__ p) {
    v16bf r;
#pragma unroll
    for (int j = 0; j < 16; ++j) r[j] = (__bf16)p[(j & 7) + ((j >> 3) << 4)];
    return r;
}

// ---------------------------------------------------------------------------
// Fused SAGEConv: out = [relu]( (agg/max(cnt,1)) @ Wl^T + bl + xdst @ Wr^T )
// One wave (32 threads) per 16 output rows. 16 WMMAs per wave
// (4 N-tiles x (2 k-steps x 2 GEMMs)).
// ---------------------------------------------------------------------------
__global__ __launch_bounds__(32)
void hg_sage_wmma(const float* __restrict__ agg,
                  const float* __restrict__ cnt,
                  const float* __restrict__ xdst,
                  const float* __restrict__ Wl,
                  const float* __restrict__ bl,
                  const float* __restrict__ Wr,
                  float* __restrict__ out,
                  int n, int do_relu) {
    const int lane = threadIdx.x;            // 0..31, wave32
    const int lm   = lane & 15;
    const bool hi  = (lane & 16) != 0;
    const int mb   = blockIdx.x * 16;

    int row = mb + lm;
    if (row >= n) row = n - 1;               // clamp loads; stores are guarded

    const float c   = cnt[row];
    const float inv = 1.0f / fmaxf(c, 1.0f);

    const float* arow = agg  + (size_t)row * HDIM;
    const float* xrow = xdst + (size_t)row * HDIM;
    const int khA = hi ? 8 : 0;

    // A operands: mean-aggregated neighbors (scaled) and self features.
    v16bf a0 = hg_pack_a(arow + 0  + khA, inv);
    v16bf a1 = hg_pack_a(arow + 32 + khA, inv);
    v16bf s0 = hg_pack_a(xrow + 0  + khA, 1.0f);
    v16bf s1 = hg_pack_a(xrow + 32 + khA, 1.0f);

    const int khB = hi ? 8 : 0;

#pragma unroll
    for (int nb = 0; nb < HDIM; nb += 16) {
        const float* wl = Wl + (size_t)(nb + lm) * HDIM + khB;
        const float* wr = Wr + (size_t)(nb + lm) * HDIM + khB;
        v16bf bl0 = hg_pack_b(wl + 0);
        v16bf bl1 = hg_pack_b(wl + 32);
        v16bf br0 = hg_pack_b(wr + 0);
        v16bf br1 = hg_pack_b(wr + 32);

        const float bias = bl[nb + lm];      // bias depends only on N = nb+lm
        v8f c8;
#pragma unroll
        for (int r = 0; r < 8; ++r) c8[r] = bias;

        c8 = __builtin_amdgcn_wmma_f32_16x16x32_bf16(false, a0, false, bl0,
                                                     (short)0, c8, false, false);
        c8 = __builtin_amdgcn_wmma_f32_16x16x32_bf16(false, a1, false, bl1,
                                                     (short)0, c8, false, false);
        c8 = __builtin_amdgcn_wmma_f32_16x16x32_bf16(false, s0, false, br0,
                                                     (short)0, c8, false, false);
        c8 = __builtin_amdgcn_wmma_f32_16x16x32_bf16(false, s1, false, br1,
                                                     (short)0, c8, false, false);

        // D layout: VGPR r -> M = r (lo lanes) / r+8 (hi lanes), N = lane&15
        const int col   = nb + lm;
        const int rbase = mb + (hi ? 8 : 0);
#pragma unroll
        for (int r = 0; r < 8; ++r) {
            const int orow = rbase + r;
            if (orow < n) {
                float v = c8[r];
                if (do_relu) v = fmaxf(v, 0.0f);
                out[(size_t)orow * HDIM + col] = v;
            }
        }
    }
}

// ---------------------------------------------------------------------------
// launch
// ---------------------------------------------------------------------------
extern "C" void kernel_launch(void* const* d_in, const int* in_sizes, int n_in,
                              void* d_out, int out_size, void* d_ws, size_t ws_size,
                              hipStream_t stream) {
    const float* x_user    = (const float*)d_in[0];
    const float* x_problem = (const float*)d_in[1];
    const float* x_topic   = (const float*)d_in[2];
    const int*   up_src = (const int*)d_in[3];
    const int*   up_dst = (const int*)d_in[4];
    const int*   pt_src = (const int*)d_in[5];
    const int*   pt_dst = (const int*)d_in[6];
    const int*   pu_src = (const int*)d_in[7];
    const int*   pu_dst = (const int*)d_in[8];
    const float* W1_up_l = (const float*)d_in[9];
    const float* W1_up_r = (const float*)d_in[10];
    const float* W1_pt_l = (const float*)d_in[11];
    const float* W1_pt_r = (const float*)d_in[12];
    const float* W1_pu_l = (const float*)d_in[13];
    const float* W1_pu_r = (const float*)d_in[14];
    const float* W2_up_l = (const float*)d_in[15];
    const float* W2_up_r = (const float*)d_in[16];
    const float* W2_pt_l = (const float*)d_in[17];
    const float* W2_pt_r = (const float*)d_in[18];
    const float* W2_pu_l = (const float*)d_in[19];
    const float* W2_pu_r = (const float*)d_in[20];
    const float* b1_up = (const float*)d_in[21];
    const float* b1_pt = (const float*)d_in[22];
    const float* b1_pu = (const float*)d_in[23];
    const float* b2_up = (const float*)d_in[24];
    const float* b2_pt = (const float*)d_in[25];
    const float* b2_pu = (const float*)d_in[26];

    const int nU = in_sizes[0] / HDIM;
    const int nP = in_sizes[1] / HDIM;
    const int nT = in_sizes[2] / HDIM;
    const int eUP = in_sizes[3];
    const int ePT = in_sizes[5];
    const int ePU = in_sizes[7];

    // workspace layout (floats): [aggP, aggT, aggU][cntP, cntT, cntU][hU, hP, hT]
    float* ws   = (float*)d_ws;
    float* aggP = ws;
    float* aggT = aggP + (size_t)nP * HDIM;
    float* aggU = aggT + (size_t)nT * HDIM;
    float* cntP = aggU + (size_t)nU * HDIM;
    float* cntT = cntP + nP;
    float* cntU = cntT + nT;
    float* hU   = cntU + nU;
    float* hP   = hU + (size_t)nU * HDIM;
    float* hT   = hP + (size_t)nP * HDIM;

    const size_t aggElems = ((size_t)nP + nT + nU) * HDIM;
    const size_t cntElems = (size_t)nP + nT + nU;

    float* oU = (float*)d_out;
    float* oP = oU + (size_t)nU * HDIM;
    float* oT = oP + (size_t)nP * HDIM;

    const int ZB = 256;
    auto sblocks = [](long long e) { return (unsigned)((e * 16 + 255) / 256); };
    auto gblocks = [](int n) { return (unsigned)((n + 15) / 16); };

    // ---- layer 1 ----
    hg_zero_f32<<<2048, ZB, 0, stream>>>(aggP, aggElems + cntElems);

    hg_scatter_add<<<sblocks(eUP), 256, 0, stream>>>(x_user,    up_src, up_dst, aggP, cntP, eUP, 1);
    hg_scatter_add<<<sblocks(ePT), 256, 0, stream>>>(x_problem, pt_src, pt_dst, aggT, cntT, ePT, 1);
    hg_scatter_add<<<sblocks(ePU), 256, 0, stream>>>(x_problem, pu_src, pu_dst, aggU, cntU, ePU, 1);

    hg_sage_wmma<<<gblocks(nP), 32, 0, stream>>>(aggP, cntP, x_problem, W1_up_l, b1_up, W1_up_r, hP, nP, 1);
    hg_sage_wmma<<<gblocks(nT), 32, 0, stream>>>(aggT, cntT, x_topic,   W1_pt_l, b1_pt, W1_pt_r, hT, nT, 1);
    hg_sage_wmma<<<gblocks(nU), 32, 0, stream>>>(aggU, cntU, x_user,    W1_pu_l, b1_pu, W1_pu_r, hU, nU, 1);

    // ---- layer 2 (in-degrees unchanged: reuse cnt*) ----
    hg_zero_f32<<<2048, ZB, 0, stream>>>(aggP, aggElems);

    hg_scatter_add<<<sblocks(eUP), 256, 0, stream>>>(hU, up_src, up_dst, aggP, cntP, eUP, 0);
    hg_scatter_add<<<sblocks(ePT), 256, 0, stream>>>(hP, pt_src, pt_dst, aggT, cntT, ePT, 0);
    hg_scatter_add<<<sblocks(ePU), 256, 0, stream>>>(hP, pu_src, pu_dst, aggU, cntU, ePU, 0);

    hg_sage_wmma<<<gblocks(nP), 32, 0, stream>>>(aggP, cntP, hP, W2_up_l, b2_up, W2_up_r, oP, nP, 0);
    hg_sage_wmma<<<gblocks(nT), 32, 0, stream>>>(aggT, cntT, hT, W2_pt_l, b2_pt, W2_pt_r, oT, nT, 0);
    hg_sage_wmma<<<gblocks(nU), 32, 0, stream>>>(aggU, cntU, hU, W2_pu_l, b2_pu, W2_pu_r, oU, nU, 0);
}
